// MultiHeadLocationAwareAttention_40398462386521
// MI455X (gfx1250) — matrix operand order repaired and checked by reading, not compiled, not run on hardware
//
#include <hip/hip_runtime.h>
#include <hip/hip_bf16.h>
#include <math.h>

// Problem dims (compile-time)
#define BDIM 8
#define QLEN 512
#define VLEN 2048
#define DMOD 1024
#define NH   8
#define DHD  128
#define KF   10

typedef __attribute__((ext_vector_type(16))) __bf16 v16bf;
typedef __attribute__((ext_vector_type(8)))  __bf16 v8bf;
typedef __attribute__((ext_vector_type(8)))  float  v8f;
typedef __attribute__((ext_vector_type(4)))  unsigned int u32x4;
typedef __attribute__((ext_vector_type(8)))  int i32x8;
typedef __attribute__((ext_vector_type(4)))  int i32x4;

// float -> bf16 (round-to-nearest-even), stored as ushort so LDS stays POD
__device__ __forceinline__ unsigned short f2bf(float f) {
    unsigned int u = __float_as_uint(f);
    return (unsigned short)((u + 0x7FFFu + ((u >> 16) & 1u)) >> 16);
}

// sigmoid via v_exp_f32 + v_rcp_f32 (avoids the IEEE divide expansion)
__device__ __forceinline__ float fast_sigmoid(float x) {
    return __builtin_amdgcn_rcpf(1.0f + __expf(-x));
}

// Build a v16bf fragment from two 16-byte LDS reads
__device__ __forceinline__ v16bf frag16(const unsigned short* p0, const unsigned short* p1) {
    v8bf lo = *(const v8bf*)p0;
    v8bf hi = *(const v8bf*)p1;
    return __builtin_shufflevector(lo, hi, 0,1,2,3,4,5,6,7,8,9,10,11,12,13,14,15);
}

__device__ __forceinline__ v8f wmma_bf16(v16bf a, v16bf b, v8f c) {
    return __builtin_amdgcn_wmma_f32_16x16x32_bf16(false, a, false, b, (short)0, c, false, false);
}

// ---------------------------------------------------------------------------
// Tensor Data Mover: 2D bf16 tile (tileX elems per row, tileY rows) from
// global (row stride = strideElems) into LDS at ldsOff, with LDS padding of
// (padAmt+1) DWORDs inserted every 2^(padIv+1) DWORDs (D# pad feature) so the
// landed tile keeps our bank-conflict-free row strides.
__device__ __forceinline__ void tdm_load_2d_bf16(unsigned ldsOff, const void* gaddr,
                                                 int tileX, int tileY,
                                                 long long strideElems,
                                                 int padIv, int padAmt) {
    unsigned long long ga = (unsigned long long)gaddr;
    u32x4 g0;
    g0.x = 1u;                                              // count=1, user D#
    g0.y = ldsOff;                                          // lds_addr (bytes)
    g0.z = (unsigned)(ga & 0xFFFFFFFFull);                  // global_addr[31:0]
    g0.w = (unsigned)((ga >> 32) & 0x01FFFFFFull) | (2u << 30); // [56:32] | type=2
    i32x8 g1;
    g1[0] = (1 << 16) | (1 << 20) | (padIv << 22) | (padAmt << 25); // data_size=2B, pad_enable
    g1[1] = (tileX & 0xFFFF) << 16;                         // tensor_dim0[15:0] @ bit48
    g1[2] = ((unsigned)tileX >> 16) | ((tileY & 0xFFFF) << 16); // dim0 hi | tensor_dim1 lo
    g1[3] = ((unsigned)tileY >> 16) | ((tileX & 0xFFFF) << 16); // dim1 hi | tile_dim0
    g1[4] = (tileY & 0xFFFF);                               // tile_dim1 (tile_dim2 = 0)
    g1[5] = (int)(strideElems & 0xFFFFFFFFll);              // tensor_dim0_stride lo32
    g1[6] = (int)((strideElems >> 32) & 0xFFFFll);          // stride hi16 (dim1_stride=0)
    g1[7] = 0;
    i32x4 z4 = {0, 0, 0, 0};
    i32x8 z8 = {0, 0, 0, 0, 0, 0, 0, 0};
    __builtin_amdgcn_tensor_load_to_lds(g0, g1, z4, z4, z8, 0);
}

// ---------------------------------------------------------------------------
// conv1d over last_align -> U (B*QL, 10)
__global__ void conv_kernel(const float* __restrict__ la, const float* __restrict__ cw,
                            const float* __restrict__ cb, float* __restrict__ U) {
    int m = blockIdx.x * blockDim.x + threadIdx.x;
    if (m >= BDIM * QLEN) return;
    int b = m >> 9, q = m & (QLEN - 1);
    float x0 = (q > 0)        ? la[b * QLEN + q - 1] : 0.f;
    float x1 =                  la[b * QLEN + q];
    float x2 = (q < QLEN - 1) ? la[b * QLEN + q + 1] : 0.f;
#pragma unroll
    for (int j = 0; j < KF; ++j)
        U[m * KF + j] = cb[j] + x0 * cw[j*3 + 0] + x1 * cw[j*3 + 1] + x2 * cw[j*3 + 2];
}

// preadd[m,n] = bias[n] + bq[n] + bu[n] + sum_j U[m,j] * Wu[n,j]
__global__ void preadd_kernel(const float* __restrict__ U, const float* __restrict__ Wu,
                              const float* __restrict__ bq, const float* __restrict__ bu,
                              const float* __restrict__ bias, float* __restrict__ pre) {
    int idx = blockIdx.x * blockDim.x + threadIdx.x;
    int m = idx >> 10, n = idx & 1023;
    float acc = bias[n] + bq[n] + bu[n];
    const float* u  = U  + m * KF;
    const float* wu = Wu + n * KF;
#pragma unroll
    for (int j = 0; j < KF; ++j) acc += u[j] * wu[j];
    pre[idx] = acc;
}

// f32 -> bf16 bulk conversion (weights), n % 4 == 0
__global__ void cvt_bf16_kernel(const float* __restrict__ in,
                                unsigned short* __restrict__ out, int n) {
    int i = (blockIdx.x * blockDim.x + threadIdx.x) * 4;
    if (i >= n) return;
    float4 f = *(const float4*)(in + i);
    out[i+0] = f2bf(f.x); out[i+1] = f2bf(f.y);
    out[i+2] = f2bf(f.z); out[i+3] = f2bf(f.w);
}

// ---------------------------------------------------------------------------
// bf16 WMMA GEMM: D = A(MxK) @ W(NxK)^T (+ epilogue by MODE). W is bf16.
//   MODE 0: += pre[m,n]; scale by 1/sqrt(128); store bf16  (Q projection;
//           q_s bf16 feeds only the score WMMA, so the attention scale folds here)
//   MODE 1: += aux[n];        store bf16   (V projection)
//   MODE 2: tanh(x + aux[n]); store f32    (output FC)
// A split at kSplit between A0 (row stride kSplit) and A1 (row stride K-kSplit).
// B tiles staged by the Tensor Data Mover; A tiles need f32->bf16 VALU staging.
template <int MODE>
__global__ __launch_bounds__(256)
void gemm_kernel(const float* __restrict__ A0, const float* __restrict__ A1, int kSplit,
                 const unsigned short* __restrict__ Bw, int Kdim,
                 const float* __restrict__ aux, const float* __restrict__ pre,
                 float* __restrict__ outF, unsigned short* __restrict__ outB, int N) {
    __shared__ unsigned short Al[128][40];   // 80B rows (16B aligned)
    __shared__ unsigned short Bl[64][40];    // TDM pad: 16B per 64B -> 80B rows
    const int tid  = threadIdx.x;
    const int lane = tid & 31, w = tid >> 5;
    const int wm = w & 3, wn = w >> 2;       // 4x2 wave grid over 128x64
    const int m0 = blockIdx.x * 128, n0 = blockIdx.y * 64;
    const int kA1 = Kdim - kSplit;
    const int khalf = lane >> 4, l16 = lane & 15;
    const bool wave0 = (__builtin_amdgcn_readfirstlane(w) == 0);

    v8f acc[2][2] = {};

    for (int k0 = 0; k0 < Kdim; k0 += 32) {
        // Kick off B tile DMA (64 rows x 32 bf16), overlaps with A staging below.
        if (wave0)
            tdm_load_2d_bf16((unsigned)(size_t)(void*)Bl,
                             Bw + (size_t)n0 * Kdim + k0, 32, 64, Kdim,
                             /*padIv=16DW*/3, /*padAmt=4DW*/3);
        // Stage A tile (128x32 f32 -> bf16)
#pragma unroll
        for (int i = 0; i < 4; ++i) {
            int idx = tid + i * 256;
            int r = idx >> 3, c = (idx & 7) * 4;
            int gk = k0 + c;
            const float* src = (gk < kSplit)
                ? (A0 + (size_t)(m0 + r) * kSplit + gk)
                : (A1 + (size_t)(m0 + r) * kA1 + (gk - kSplit));
            float4 f = *(const float4*)src;
            Al[r][c+0] = f2bf(f.x); Al[r][c+1] = f2bf(f.y);
            Al[r][c+2] = f2bf(f.z); Al[r][c+3] = f2bf(f.w);
        }
        if (k0 + 32 < Kdim) {  // pull next A tile toward L2 (global_prefetch_b8)
            int gk = k0 + 32;
            const float* pf = (gk < kSplit)
                ? (A0 + (size_t)(m0 + (tid >> 1)) * kSplit + gk)
                : (A1 + (size_t)(m0 + (tid >> 1)) * kA1 + (gk - kSplit));
            __builtin_prefetch(pf, 0, 1);
        }
        if (wave0) __builtin_amdgcn_s_wait_tensorcnt(0);
        __syncthreads();

        v16bf afr[2], bfr[2];
#pragma unroll
        for (int mi = 0; mi < 2; ++mi) {
            const unsigned short* row = Al[wm*32 + mi*16 + l16];
            afr[mi] = frag16(row + khalf*8, row + 16 + khalf*8);
        }
#pragma unroll
        for (int ni = 0; ni < 2; ++ni) {
            const unsigned short* row = Bl[wn*32 + ni*16 + l16];
            bfr[ni] = frag16(row + khalf*8, row + 16 + khalf*8);
        }
#pragma unroll
        for (int mi = 0; mi < 2; ++mi)
#pragma unroll
            for (int ni = 0; ni < 2; ++ni)
                acc[mi][ni] = wmma_bf16(afr[mi], bfr[ni], acc[mi][ni]);
        __syncthreads();
    }

#pragma unroll
    for (int mi = 0; mi < 2; ++mi)
#pragma unroll
        for (int ni = 0; ni < 2; ++ni)
#pragma unroll
            for (int v = 0; v < 8; ++v) {
                int ml = m0 + wm*32 + mi*16 + khalf*8 + v;
                int nl = n0 + wn*32 + ni*16 + l16;
                size_t o = (size_t)ml * N + nl;
                float val = acc[mi][ni][v];
                if (MODE == 0) {      // fold attention scale 1/sqrt(128)
                    val = (val + pre[o]) * 0.08838834764831845f;
                    outB[o] = f2bf(val);
                } else if (MODE == 1) {
                    val += aux[nl];
                    outB[o] = f2bf(val);
                } else {
                    val = tanhf(val + aux[nl]);
                    outF[o] = val;
                }
            }
}

// ---------------------------------------------------------------------------
// Attention: one block per (b, h, 64-row q-tile). Two passes over VL.
//   pass A: rowsum of sigmoid(scores), V tiles via double-buffered TDM DMA.
//   pass B: recompute, write normalized align once, context WMMA (no renorm pass).
// LDS (dynamic, 80128 B):
//   Qs  [64][136]  @ 0        (17408)   TDM pad keeps 272B row stride
//   Vs0 [64][136]  @ 17408    (17408)
//   Vs1 [64][136]  @ 34816    (17408)
//   VsT [128][72]  @ 52224    (18432)   dh-major copy for context B-operand
//   Sb  [64][72]   @ 70656    (9216)
//   rs  [64]       @ 79872    (256)
__global__ __launch_bounds__(256)
void attn_kernel(const unsigned short* __restrict__ qsb,  // (B*QL, D) bf16, pre-scaled
                 const unsigned short* __restrict__ vsb,  // (B*VL, D) bf16
                 float* __restrict__ alignOut,            // (H*B, QL, VL)
                 float* __restrict__ ctx) {               // (B*QL, D) f32
    extern __shared__ char smem[];
    unsigned short (*Qs)[136]  = (unsigned short (*)[136])(smem + 0);
    unsigned short (*Vs0)[136] = (unsigned short (*)[136])(smem + 17408);
    unsigned short (*Vs1)[136] = (unsigned short (*)[136])(smem + 34816);
    unsigned short (*VsT)[72]  = (unsigned short (*)[72])(smem + 52224);
    unsigned short (*Sb)[72]   = (unsigned short (*)[72])(smem + 70656);
    float* rs = (float*)(smem + 79872);

    const int tid = threadIdx.x, lane = tid & 31, w = tid >> 5;
    const int qt = blockIdx.x & 7;
    const int h  = (blockIdx.x >> 3) & 7;
    const int b  = blockIdx.x >> 6;
    const int q0 = qt * 64;
    const int wm = w & 3, wq = w >> 2;
    const int khalf = lane >> 4, l16 = lane & 15;
    const bool wave0 = (__builtin_amdgcn_readfirstlane(w) == 0);
    const int NC = VLEN / 64;

    const unsigned short* qsrc  = qsb + (size_t)(b*QLEN + q0) * DMOD + h*DHD;
    const unsigned short* vbase = vsb + (size_t)(b*VLEN) * DMOD + h*DHD;

    if (tid < 64) rs[tid] = 0.f;
    if (wave0) {   // DMA Q tile + first V chunk (pad -> 272B rows)
        tdm_load_2d_bf16((unsigned)(size_t)(void*)Qs,  qsrc,  128, 64, DMOD, 5, 3);
        tdm_load_2d_bf16((unsigned)(size_t)(void*)Vs0, vbase, 128, 64, DMOD, 5, 3);
    }

    // ----- pass A: row sums of sigmoid(score) -----
    for (int vc = 0; vc < NC; ++vc) {
        if (wave0) __builtin_amdgcn_s_wait_tensorcnt(0);
        __syncthreads();
        if (wave0 && vc + 1 < NC)   // prefetch next chunk into the other buffer
            tdm_load_2d_bf16((unsigned)(size_t)(void*)((vc & 1) ? Vs0 : Vs1),
                             vbase + (size_t)(vc + 1) * 64 * DMOD, 128, 64, DMOD, 5, 3);
        unsigned short (*Vc)[136] = (vc & 1) ? Vs1 : Vs0;

        v8f sacc[2] = {};
#pragma unroll
        for (int dh0 = 0; dh0 < DHD; dh0 += 32) {
            const unsigned short* ar = &Qs[wm*16 + l16][dh0];
            v16bf a = frag16(ar + khalf*8, ar + 16 + khalf*8);
#pragma unroll
            for (int j = 0; j < 2; ++j) {
                const unsigned short* br = &Vc[(wq*2 + j)*16 + l16][dh0];
                v16bf bb = frag16(br + khalf*8, br + 16 + khalf*8);
                sacc[j] = wmma_bf16(a, bb, sacc[j]);
            }
        }
#pragma unroll
        for (int j = 0; j < 2; ++j)
#pragma unroll
            for (int v = 0; v < 8; ++v) {
                int ml = wm*16 + khalf*8 + v;
                atomicAdd(&rs[ml], fast_sigmoid(sacc[j][v]));
            }
        __syncthreads();
    }
    if (tid < 64) rs[tid] = __builtin_amdgcn_rcpf(rs[tid]);  // reciprocal row sums
    if (wave0)    // restart DMA pipeline for pass B
        tdm_load_2d_bf16((unsigned)(size_t)(void*)Vs0, vbase, 128, 64, DMOD, 5, 3);

    // ----- pass B: normalized align + context accumulation -----
    v8f cacc[4] = {};
    for (int vc = 0; vc < NC; ++vc) {
        if (wave0) __builtin_amdgcn_s_wait_tensorcnt(0);
        __syncthreads();          // buffer vc ready (and rs ready on first iter)
        if (wave0 && vc + 1 < NC)
            tdm_load_2d_bf16((unsigned)(size_t)(void*)((vc & 1) ? Vs0 : Vs1),
                             vbase + (size_t)(vc + 1) * 64 * DMOD, 128, 64, DMOD, 5, 3);
        unsigned short (*Vc)[136] = (vc & 1) ? Vs1 : Vs0;

        // Build dh-major copy for the context B-operand (LDS -> LDS)
#pragma unroll
        for (int i = 0; i < 4; ++i) {
            int idx = tid + i * 256;
            int r = idx >> 4, c = (idx & 15) * 8;
            uint4 d = *(const uint4*)&Vc[r][c];
            const unsigned short* e = (const unsigned short*)&d;
#pragma unroll
            for (int t = 0; t < 8; ++t) VsT[c + t][r] = e[t];
        }
        __syncthreads();

        v8f sacc[2] = {};
#pragma unroll
        for (int dh0 = 0; dh0 < DHD; dh0 += 32) {
            const unsigned short* ar = &Qs[wm*16 + l16][dh0];
            v16bf a = frag16(ar + khalf*8, ar + 16 + khalf*8);
#pragma unroll
            for (int j = 0; j < 2; ++j) {
                const unsigned short* br = &Vc[(wq*2 + j)*16 + l16][dh0];
                v16bf bb = frag16(br + khalf*8, br + 16 + khalf*8);
                sacc[j] = wmma_bf16(a, bb, sacc[j]);
            }
        }
#pragma unroll
        for (int j = 0; j < 2; ++j) {
            int nl = (wq*2 + j)*16 + l16;
#pragma unroll
            for (int v = 0; v < 8; ++v) {
                int ml = wm*16 + khalf*8 + v;
                float sn = fast_sigmoid(sacc[j][v]) * rs[ml];   // normalized prob
                alignOut[((size_t)((h*BDIM + b)*QLEN + q0 + ml)) * VLEN + vc*64 + nl] = sn;
                Sb[ml][nl] = f2bf(sn);
            }
        }
        __syncthreads();

        // context += Sn(64x64) @ V(64x128)
#pragma unroll
        for (int ks = 0; ks < 64; ks += 32) {
            const unsigned short* ar = &Sb[wm*16 + l16][ks];
            v16bf a = frag16(ar + khalf*8, ar + 16 + khalf*8);
#pragma unroll
            for (int j = 0; j < 4; ++j) {
                const unsigned short* br = &VsT[(wq*4 + j)*16 + l16][ks];
                v16bf bb = frag16(br + khalf*8, br + 16 + khalf*8);
                cacc[j] = wmma_bf16(a, bb, cacc[j]);
            }
        }
        __syncthreads();
    }

    // store context (already normalized)
#pragma unroll
    for (int j = 0; j < 4; ++j) {
        int nl = (wq*4 + j)*16 + l16;
#pragma unroll
        for (int v = 0; v < 8; ++v) {
            int ml = wm*16 + khalf*8 + v;
            ctx[((size_t)(b*QLEN + q0 + ml)) * DMOD + h*DHD + nl] = cacc[j][v];
        }
    }
}

// ---------------------------------------------------------------------------
extern "C" void kernel_launch(void* const* d_in, const int* in_sizes, int n_in,
                              void* d_out, int out_size, void* d_ws, size_t ws_size,
                              hipStream_t stream) {
    const float* Q    = (const float*)d_in[0];
    const float* V    = (const float*)d_in[1];
    const float* la   = (const float*)d_in[2];
    const float* cw   = (const float*)d_in[3];
    const float* cb   = (const float*)d_in[4];
    const float* Wq   = (const float*)d_in[5];
    const float* bq   = (const float*)d_in[6];
    const float* Wv   = (const float*)d_in[7];
    const float* bv   = (const float*)d_in[8];
    const float* Wu   = (const float*)d_in[9];
    const float* bu   = (const float*)d_in[10];
    const float* bias = (const float*)d_in[11];
    const float* fcw  = (const float*)d_in[12];
    const float* fcb  = (const float*)d_in[13];

    char* ws = (char*)d_ws;
    float*          U    = (float*)(ws);                          // 160 KB
    float*          pre  = (float*)(ws + (1ull  << 20));          // 16 MB
    unsigned short* qsb  = (unsigned short*)(ws + (18ull << 20)); // 8 MB bf16
    unsigned short* vsb  = (unsigned short*)(ws + (27ull << 20)); // 32 MB bf16
    float*          ctx  = (float*)(ws + (60ull << 20));          // 16 MB
    unsigned short* wqb  = (unsigned short*)(ws + (77ull << 20)); // 2 MB bf16
    unsigned short* wvb  = (unsigned short*)(ws + (80ull << 20)); // 2 MB bf16
    unsigned short* fwb  = (unsigned short*)(ws + (83ull << 20)); // 4 MB bf16

    float* outO     = (float*)d_out;                              // (B,QL,D)
    float* alignOut = outO + (size_t)BDIM * QLEN * DMOD;          // (H*B,QL,VL)

    (void)hipFuncSetAttribute((const void*)attn_kernel,
                              hipFuncAttributeMaxDynamicSharedMemorySize, 80128);

    conv_kernel<<<(BDIM*QLEN + 255) / 256, 256, 0, stream>>>(la, cw, cb, U);
    preadd_kernel<<<(BDIM*QLEN*DMOD) / 256, 256, 0, stream>>>(U, Wu, bq, bu, bias, pre);

    cvt_bf16_kernel<<<(DMOD*DMOD/4 + 255) / 256, 256, 0, stream>>>(Wq, wqb, DMOD*DMOD);
    cvt_bf16_kernel<<<(DMOD*DMOD/4 + 255) / 256, 256, 0, stream>>>(Wv, wvb, DMOD*DMOD);
    cvt_bf16_kernel<<<(2*DMOD*DMOD/4 + 255) / 256, 256, 0, stream>>>(fcw, fwb, 2*DMOD*DMOD);

    dim3 gq(BDIM*QLEN / 128, DMOD / 64);          // 32 x 16
    gemm_kernel<0><<<gq, 256, 0, stream>>>(Q, Q, DMOD, wqb, DMOD,
                                           nullptr, pre, nullptr, qsb, DMOD);
    dim3 gv(BDIM*VLEN / 128, DMOD / 64);          // 128 x 16
    gemm_kernel<1><<<gv, 256, 0, stream>>>(V, V, DMOD, wvb, DMOD,
                                           bv, nullptr, nullptr, vsb, DMOD);

    attn_kernel<<<BDIM * NH * (QLEN / 64), 256, 80128, stream>>>(qsb, vsb, alignOut, ctx);

    dim3 gf(BDIM*QLEN / 128, DMOD / 64);          // K = 2048, split ctx|Q at 1024
    gemm_kernel<2><<<gf, 256, 0, stream>>>(ctx, Q, DMOD, fwb, 2*DMOD,
                                           fcb, nullptr, outO, nullptr, DMOD);
}
